// qrnn5_42760694399687
// MI455X (gfx1250) — matrix operand order repaired and checked by reading, not compiled, not run on hardware
//
#include <hip/hip_runtime.h>

typedef __attribute__((ext_vector_type(16))) _Float16 v16h;
typedef __attribute__((ext_vector_type(4)))  _Float16 v4h;
typedef __attribute__((ext_vector_type(8)))  float    v8f;

#define VOCAB 32000
#define E_DIM 256
#define H_DIM 256
#define B_DIM 64
#define T_DIM 2048

// d_out[b] = b_out (atomicAdd target), re-initialized every call (deterministic).
__global__ void qrnn_init_out(float* __restrict__ out, const float* __restrict__ b_out) {
  int b = threadIdx.x;
  if (b < B_DIM) out[b] = b_out[0];
}

// One wave (32 lanes) per block; block owns a 16(b) x 16(h) tile of the recurrence.
// grid = (H/16, B/16) = (16, 4) -> 64 persistent blocks, each loops t = 0..T-1.
__global__ __launch_bounds__(32)
void qrnn_fused(const int* __restrict__ X,
                const float* __restrict__ emb,
                const float* __restrict__ W_qrnn,
                const float* __restrict__ b_qrnn,
                const float* __restrict__ c0,
                const float* __restrict__ W_out,
                float* __restrict__ out) {
  // A fragments: 8 k-chunks x 32 lanes x 16 halves (contiguous k = 16*half + s order used here)
  __shared__ __attribute__((aligned(32))) _Float16 sA[8 * 32 * 16];        // 8 KB
  // B fragments for z, f, o column groups (staging; z/f promoted to VGPRs)
  __shared__ __attribute__((aligned(32))) _Float16 sW[3 * 8 * 32 * 16];    // 24 KB

  const int lane = threadIdx.x;
  const int h0 = blockIdx.x * 16;
  const int b0 = blockIdx.y * 16;

  // ---- one-time: swizzle W_qrnn columns [h0..h0+15] of z/f/o into B-fragment order ----
  // B 32x16 f16, contiguous k order used here: lane n<16 holds K=chunk*32+(0..15) of column n; lane n+16 holds K+16..31.
  for (int e = lane; e < 3 * 16 * 256; e += 32) {
    int mat = e >> 12;            // 0=z, 1=f, 2=o  (4096 elements each)
    int rem = e & 4095;
    int k   = rem >> 4;
    int n   = rem & 15;
    float w = W_qrnn[k * (3 * H_DIM) + mat * H_DIM + h0 + n];
    int chunk = k >> 5, kk = k & 31;
    int bl  = n + ((kk >> 4) << 4);
    sW[mat * 4096 + chunk * 512 + bl * 16 + (kk & 15)] = (_Float16)w;
  }
  __syncthreads();

  // ---- hoist loop-invariant z/f B-fragments into VGPRs (128 VGPRs, fine in wave32) ----
  v16h wz[8], wf[8];
#pragma unroll
  for (int chunk = 0; chunk < 8; ++chunk) {
    wz[chunk] = *(const v16h*)&sW[0 * 4096 + chunk * 512 + lane * 16];
    wf[chunk] = *(const v16h*)&sW[1 * 4096 + chunk * 512 + lane * 16];
  }

  // Per-lane constants. C/D f32 16x16 layout: lane l -> N = l&15, M = r + 8*(l>>4).
  const int nIdx  = lane & 15;
  const int mBase = (lane >> 4) << 3;
  const float bz_s   = b_qrnn[h0 + nIdx];
  const float bf_s   = b_qrnn[H_DIM + h0 + nIdx];
  const float bo_s   = b_qrnn[2 * H_DIM + h0 + nIdx];
  const float wout_s = W_out[h0 + nIdx];

  // Recurrent state lives in VGPRs in exactly the WMMA accumulator layout.
  float c[8];
#pragma unroll
  for (int r = 0; r < 8; ++r)
    c[r] = c0[(b0 + mBase + r) * H_DIM + h0 + nIdx];

  const int gm    = lane >> 1;   // which b-row this lane gathers (2 lanes per row)
  const int ghalf = lane & 1;    // which 128-float half of the row

  v8f acc_o = {};
  for (int t = 0; t < T_DIM; ++t) {
    // ---- gather emb rows for this timestep into LDS, pre-swizzled to A-fragment order ----
    const int row = X[(b0 + gm) * T_DIM + t];
    const float4* src = (const float4*)(emb + (size_t)row * E_DIM) + ghalf * 32;
#pragma unroll 8
    for (int j = 0; j < 32; ++j) {
      float4 v = src[j];
      int k = ghalf * 128 + j * 4;
      v4h p = {(_Float16)v.x, (_Float16)v.y, (_Float16)v.z, (_Float16)v.w};
      // A 16x32 f16, contiguous k order used here (same as B): lane m holds K 0..15, lane m+16 holds K 16..31
      int chunk = k >> 5, sub = (k >> 3) & 3;
      int al  = gm + ((sub >> 1) << 4);
      int idx = ((sub & 1) << 3) | (k & 7);
      *(v4h*)&sA[chunk * 512 + al * 16 + idx] = p;   // 8B aligned -> ds_store_b64
    }
    if (t + 1 < T_DIM) {
      int nrow = X[(b0 + gm) * T_DIM + t + 1];
      __builtin_prefetch(emb + (size_t)nrow * E_DIM + ghalf * 128, 0, 0); // global_prefetch_b8
    }
    __syncthreads();

    // ---- z/f tiles: D = A(16x256) x B(256x16); B operands already in VGPRs ----
    v8f az = {}, af = {};
#pragma unroll
    for (int chunk = 0; chunk < 8; ++chunk) {
      v16h a = *(const v16h*)&sA[chunk * 512 + lane * 16];
      az = __builtin_amdgcn_wmma_f32_16x16x32_f16(false, a, false, wz[chunk], (short)0, az, false, false);
      af = __builtin_amdgcn_wmma_f32_16x16x32_f16(false, a, false, wf[chunk], (short)0, af, false, false);
    }
    // o-gate only matters at the final timestep (uniform branch, EXEC stays all-ones).
    if (t == T_DIM - 1) {
#pragma unroll
      for (int chunk = 0; chunk < 8; ++chunk) {
        v16h a  = *(const v16h*)&sA[chunk * 512 + lane * 16];
        v16h bo = *(const v16h*)&sW[2 * 4096 + chunk * 512 + lane * 16];
        acc_o = __builtin_amdgcn_wmma_f32_16x16x32_f16(false, a, false, bo, (short)0, acc_o, false, false);
      }
    }

    // ---- fo-pooling: c = f*c + (1-f)*z, state stays in registers ----
#pragma unroll
    for (int r = 0; r < 8; ++r) {
      float z = tanhf(az[r] + bz_s);
      float f = 1.0f / (1.0f + __expf(-(af[r] + bf_s)));
      c[r] = f * c[r] + (1.0f - f) * z;
    }
    __syncthreads();   // sA reused next iteration
  }

  // ---- output: h = sigmoid(o) * c; out[b] += sum_h h * W_out[h] ----
  float s[8];
#pragma unroll
  for (int r = 0; r < 8; ++r) {
    float o = 1.0f / (1.0f + __expf(-(acc_o[r] + bo_s)));
    s[r] = o * c[r] * wout_s;
  }
  // reduce across the N dimension (lanes 0..15 and 16..31 independently; xor<16 never crosses bit 4)
#pragma unroll
  for (int m = 1; m < 16; m <<= 1) {
#pragma unroll
    for (int r = 0; r < 8; ++r) s[r] += __shfl_xor(s[r], m, 32);
  }
  if (nIdx == 0) {   // lane 0 -> rows b0..b0+7, lane 16 -> rows b0+8..b0+15
#pragma unroll
    for (int r = 0; r < 8; ++r)
      atomicAdd(&out[b0 + mBase + r], s[r]);
  }
}

extern "C" void kernel_launch(void* const* d_in, const int* in_sizes, int n_in,
                              void* d_out, int out_size, void* d_ws, size_t ws_size,
                              hipStream_t stream) {
  (void)in_sizes; (void)n_in; (void)out_size; (void)d_ws; (void)ws_size;
  const int*   X      = (const int*)  d_in[0];
  const float* emb    = (const float*)d_in[1];
  const float* W_qrnn = (const float*)d_in[2];
  const float* b_qrnn = (const float*)d_in[3];
  const float* c0     = (const float*)d_in[4];
  const float* W_out  = (const float*)d_in[5];
  const float* b_out  = (const float*)d_in[6];
  float* out = (float*)d_out;

  qrnn_init_out<<<1, 64, 0, stream>>>(out, b_out);
  dim3 grid(H_DIM / 16, B_DIM / 16);   // 16 h-tiles x 4 b-tiles = 64 blocks
  qrnn_fused<<<grid, 32, 0, stream>>>(X, emb, W_qrnn, b_qrnn, c0, W_out, out);
}
